// SingleChannelInterp_15547781611865
// MI455X (gfx1250) — matrix operand order repaired and checked by reading, not compiled, not run on hardware
//
#include <hip/hip_runtime.h>
#include <hip/hip_bf16.h>

typedef __attribute__((ext_vector_type(16))) _Float16 v16h;
typedef __attribute__((ext_vector_type(8)))  float    v8f;

namespace {
constexpr int B_ = 16, T_ = 512, R_ = 128, V_ = 64;
constexpr int BRV_ = B_ * R_ * V_;
}

// One wave handles one (b, v) pair; block = 8 waves = 8 consecutive v.
// grid = (8 v-groups, 16 b, 4 r-tile-groups); each block does 2 r-tiles of 16.
// Reduction over T is done with v_wmma_f32_16x16x32_f16:
//   A[16r x 32t] = exp(-pk*d2)  (lp)  or  E^10 (hp, via multiplies)
//   B[32t x 16n] : col0 = m, col1 = m*x, cols 2..15 = 0
//   C/D f32 16x16 accumulate over the 16 k-steps (T=512 = 16*32).
__global__ __launch_bounds__(256)
void interp_wmma_kernel(const float* __restrict__ x,
                        const float* __restrict__ m,
                        const float* __restrict__ tt,
                        const float* __restrict__ kern,
                        const float* __restrict__ ref,
                        float* __restrict__ out)
{
  __shared__ float    sh_t[T_];          // 2 KB
  __shared__ _Float16 sh_m [8][T_];      // 8 KB
  __shared__ _Float16 sh_mx[8][T_];      // 8 KB

  const int tid   = threadIdx.x;
  const int lane  = tid & 31;
  const int wave  = tid >> 5;
  const int vgrp  = blockIdx.x;          // 0..7
  const int b     = blockIdx.y;          // 0..15
  const int rtg   = blockIdx.z;          // 0..3
  const int vbase = vgrp * 8;
  const int v     = vbase + wave;

  // Warm the L2/WGP$ path for this block's data slab (gfx1250 global_prefetch_b8).
  __builtin_prefetch(&m[(size_t)(b * T_) * V_ + vbase], 0, 0);
  __builtin_prefetch(&x[(size_t)(b * T_) * V_ + vbase], 0, 0);

  // Stage t[b,:] (fully coalesced).
  for (int i = tid; i < T_; i += 256) sh_t[i] = tt[b * T_ + i];

  // Stage m and m*x columns for this block's 8 v's (32B-coalesced groups of 8).
  for (int idx = tid; idx < T_ * 8; idx += 256) {
    const int t_i = idx >> 3, vv = idx & 7;
    const int g   = (b * T_ + t_i) * V_ + vbase + vv;
    const float mm = m[g];
    const float xx = x[g];
    sh_m [vv][t_i] = (_Float16)mm;
    sh_mx[vv][t_i] = (_Float16)(mm * xx);
  }
  __syncthreads();

  const float pk = log1pf(__expf(kern[v]));   // softplus(kernel[v])

  const int n  = lane & 15;                   // A row / C column index
  const int hf = lane >> 4;                   // lane half
  const _Float16* __restrict__ bsrc = (n == 0) ? sh_m[wave] : sh_mx[wave];
  const bool bactive = (n < 2);

  float* __restrict__ out_sigma  = out;
  float* __restrict__ out_lambda = out + BRV_;
  float* __restrict__ out_gamma  = out + 2 * BRV_;

  for (int rti = 0; rti < 2; ++rti) {
    const int r0 = (rtg * 2 + rti) * 16;
    const float refv = ref[r0 + n];           // A row M = lane&15 -> r = r0 + M
    v8f acc_lp = {};
    v8f acc_hp = {};

    for (int ks = 0; ks < 16; ++ks) {
      const int t0 = ks * 32;

      // ---- A fragments (16-bit A 16x32 layout):
      //   elem e   (e<8) -> K = 8*hf + e
      //   elem e+8       -> K = 16 + 8*hf + e
      v16h a_lp, a_hp;
      const int ta = t0 + 8 * hf;
#pragma unroll
      for (int e = 0; e < 8; ++e) {
        const float d0 = sh_t[ta + e]      - refv;
        const float d1 = sh_t[ta + 16 + e] - refv;
        const float w0 = pk * d0 * d0;
        const float w1 = pk * d1 * d1;
        const float E0 = __expf(-w0);
        const float E1 = __expf(-w1);
        const float e20 = E0 * E0, e40 = e20 * e20;
        const float e21 = E1 * E1, e41 = e21 * e21;
        const float q0 = e40 * e40 * e20;     // E0^10
        const float q1 = e41 * e41 * e21;     // E1^10
        a_lp[e]     = (_Float16)E0;
        a_lp[e + 8] = (_Float16)E1;
        a_hp[e]     = (_Float16)q0;
        a_hp[e + 8] = (_Float16)q1;
      }

      // ---- B fragment (16-bit B 32x16 layout): elem e (e<8) -> K = 8*hf + e, elem e+8 -> K = 16 + 8*hf + e
      v16h bf;
      const int tb = t0 + 8 * hf;
#pragma unroll
      for (int e = 0; e < 16; ++e) {
        const _Float16 val = bsrc[tb + (e < 8 ? e : e + 8)];
        bf[e] = bactive ? val : (_Float16)0;
      }

      // (neg_a, A, neg_b, B, c_mod, C, reuse_a, reuse_b)
      acc_lp = __builtin_amdgcn_wmma_f32_16x16x32_f16(
          false, a_lp, false, bf, (short)0, acc_lp, false, false);
      acc_hp = __builtin_amdgcn_wmma_f32_16x16x32_f16(
          false, a_hp, false, bf, (short)0, acc_hp, false, false);
    }

    // ---- Epilogue: C layout -> lane l holds column n=l&15, rows j + 8*hf.
    // Column 0 = sum(E*m) (lambda), column 1 = sum(E*m*x).
    const int base = lane & 16;
#pragma unroll
    for (int j = 0; j < 8; ++j) {
      const float lam  = __shfl(acc_lp[j], base + 0, 32);
      const float num  = __shfl(acc_lp[j], base + 1, 32);
      const float lamh = __shfl(acc_hp[j], base + 0, 32);
      const float numh = __shfl(acc_hp[j], base + 1, 32);
      if (n == 0) {
        const int r = r0 + j + 8 * hf;
        const int o = (b * R_ + r) * V_ + v;
        out_sigma [o] = num  / fmaxf(lam,  1.0f);
        out_lambda[o] = lam;
        out_gamma [o] = numh / fmaxf(lamh, 1.0f);
      }
    }
  }
}

extern "C" void kernel_launch(void* const* d_in, const int* in_sizes, int n_in,
                              void* d_out, int out_size, void* d_ws, size_t ws_size,
                              hipStream_t stream) {
  const float* x    = (const float*)d_in[0];
  const float* m    = (const float*)d_in[1];
  const float* t    = (const float*)d_in[2];
  // d_in[3] = h : unused by the reference computation
  const float* kern = (const float*)d_in[4];
  const float* ref  = (const float*)d_in[5];
  float* out = (float*)d_out;

  dim3 grid(8, 16, 4);   // v-groups x batch x r-tile-groups
  interp_wmma_kernel<<<grid, 256, 0, stream>>>(x, m, t, kern, ref, out);
}
